// Seq2Seq_Attn_GC_GRU_41583873360306
// MI455X (gfx1250) — matrix operand order repaired and checked by reading, not compiled, not run on hardware
//
#include <hip/hip_runtime.h>

#define NODES 32000
#define NB 16
#define NN 2000
#define THIST 24
#define TFCST 24
#define HID 64
#define NF 8
#define NEDGE (16 * 2000 * 32)  // 1,024,000
#define WPACK_HALFS (12 * 2 * 32 * 16)  // 12288 halfs = 24,576 bytes

typedef __attribute__((ext_vector_type(16))) _Float16 v16h;
typedef __attribute__((ext_vector_type(8)))  _Float16 v8h;
typedef __attribute__((ext_vector_type(8)))  float    v8f;
typedef int awvec __attribute__((vector_size(16)));  // b128 payload type for async-to-LDS

#if __has_builtin(__builtin_amdgcn_global_load_async_to_lds_b128) && \
    __has_builtin(__builtin_amdgcn_s_wait_asynccnt)
#define USE_ASYNC_LDS 1
#else
#define USE_ASYNC_LDS 0
#endif

__device__ __forceinline__ float sigmoidf_(float x) { return 1.0f / (1.0f + __expf(-x)); }

// ---------------------------------------------------------------- utilities
__global__ void zero_f32_kernel(float* __restrict__ p, int n) {
  int i = blockIdx.x * blockDim.x + threadIdx.x;
  if (i < n) p[i] = 0.0f;
}

// Pack w_hh (192x64 row-major, [n][k]) into WMMA B-fragment order:
// frag(ct,ks): lane l<16 -> col n=ct*16+l, k = ks*32 + 0..15 ; lane>=16 -> k = ks*32+16..31
__global__ void pack_whh_kernel(const float* __restrict__ w, _Float16* __restrict__ pk) {
  int i = blockIdx.x * blockDim.x + threadIdx.x;  // 12288 total
  if (i >= WPACK_HALFS) return;
  int e = i & 15;
  int lane = (i >> 4) & 31;
  int ks = (i >> 9) & 1;
  int ct = i >> 10;
  int n = ct * 16 + (lane & 15);
  int k = ks * 32 + (lane >> 4) * 16 + e;
  pk[i] = (_Float16)w[n * HID + k];
}

// ---------------------------------------------------------------- graph prep
__global__ void deg_kernel(const int* __restrict__ ei, float* __restrict__ deg) {
  int e = blockIdx.x * blockDim.x + threadIdx.x;
  if (e < NEDGE) atomicAdd(&deg[ei[NEDGE + e]], 1.0f);
}

__global__ void dis_kernel(float* __restrict__ deg) {
  int i = blockIdx.x * blockDim.x + threadIdx.x;
  if (i < NODES) {
    float d = deg[i];
    deg[i] = (d > 0.0f) ? rsqrtf(fmaxf(d, 1.0f)) : 0.0f;
  }
}

__global__ void norm_kernel(const int* __restrict__ ei, const float* __restrict__ dis,
                            float* __restrict__ nrm) {
  int e = blockIdx.x * blockDim.x + threadIdx.x;
  if (e < NEDGE) nrm[e] = -(dis[ei[e]] * dis[ei[NEDGE + e]]);
}

// ---------------------------------------------------------------- cheb conv
__global__ void build_x_kernel(const float* __restrict__ xn, const float* __restrict__ feat,
                               int t, float* __restrict__ xb) {
  int m = blockIdx.x * blockDim.x + threadIdx.x;
  if (m >= NODES) return;
  int b = m / NN, n = m % NN;
  xb[m * 9] = xn[m];
  const float* fp = feat + (((size_t)b * (THIST + TFCST) + (THIST + t)) * NN + n) * NF;
#pragma unroll
  for (int f = 0; f < NF; ++f) xb[m * 9 + 1 + f] = fp[f];
}

__global__ void cheb_scatter_kernel(const int* __restrict__ ei, const float* __restrict__ nrm,
                                    const float* __restrict__ xb, float* __restrict__ tx1) {
  int e = blockIdx.x * blockDim.x + threadIdx.x;
  if (e >= NEDGE) return;
  int s = ei[e], d = ei[NEDGE + e];
  float w = nrm[e];
#pragma unroll
  for (int c = 0; c < 9; ++c) atomicAdd(&tx1[d * 9 + c], w * xb[s * 9 + c]);
}

__global__ void cheb_finish_kernel(const float* __restrict__ xb, const float* __restrict__ tx1,
                                   const float* __restrict__ w0, const float* __restrict__ w1,
                                   const float* __restrict__ bb, float* __restrict__ x2) {
  int m = blockIdx.x * blockDim.x + threadIdx.x;
  if (m >= NODES) return;
  float s = bb[0];
#pragma unroll
  for (int c = 0; c < 9; ++c) s += xb[m * 9 + c] * w0[c] + tx1[m * 9 + c] * w1[c];
  float xg = sigmoidf_(s);
#pragma unroll
  for (int c = 0; c < 9; ++c) x2[m * 10 + c] = xb[m * 9 + c];
  x2[m * 10 + 9] = xg;
}

// ---------------------------------------------------------------- GRU (WMMA)
// One wave computes gh = h(16x64) @ Whh^T(64x192) via 12 column tiles x 2 k-steps
// of v_wmma_f32_16x16x32_f16, then fuses the full GRU gate math in the epilogue.
// The packed w_hh (24 KB) is staged once per workgroup into LDS — via the gfx1250
// async-to-LDS path when available — so the 4 waves share one copy.
template <int D, bool ENC>
__global__ __launch_bounds__(128) void gru_step_kernel(
    const _Float16* __restrict__ h16_in, const float* __restrict__ h32_in,
    float* __restrict__ h32_out, _Float16* __restrict__ h16_out,
    const float* __restrict__ xin, const float* __restrict__ pm25, int t,
    const _Float16* __restrict__ wpack, const float* __restrict__ w_ih,
    const float* __restrict__ b_ih, const float* __restrict__ b_hh,
    _Float16* __restrict__ hist) {
  __shared__ _Float16 wlds[WPACK_HALFS];
  const int lane = threadIdx.x & 31;
  const int wv = threadIdx.x >> 5;
  const int m0 = (blockIdx.x * 4 + wv) * 16;

  // ---- stage packed weights into LDS (1536 x b128, 12 iters of 128 threads)
#if USE_ASYNC_LDS
#pragma unroll
  for (int i = 0; i < 12; ++i) {
    const int idx = (i * 128 + (int)threadIdx.x) * 8;
    __builtin_amdgcn_global_load_async_to_lds_b128(
        (__attribute__((address_space(1))) awvec*)(wpack + idx),
        (__attribute__((address_space(3))) awvec*)(wlds + idx), 0, 0);
  }
  __builtin_amdgcn_s_wait_asynccnt(0);
  __syncthreads();
#else
#pragma unroll
  for (int i = 0; i < 12; ++i) {
    const int idx = (i * 128 + (int)threadIdx.x) * 8;
    *(v8h*)(wlds + idx) = *(const v8h*)(wpack + idx);
  }
  __syncthreads();
#endif

  // --- A fragments (16x32 f16): lane<16 holds row m0+lane, K={kk+0..15}; lane>=16 K={kk+16..31}
  const int mrow = m0 + (lane & 15);
  const int koff = (lane < 16) ? 0 : 16;
  const _Float16* hrow = h16_in + mrow * HID;
  v8h a0l = *(const v8h*)(hrow + koff);
  v8h a0h = *(const v8h*)(hrow + 8 + koff);
  v8h a1l = *(const v8h*)(hrow + 32 + koff);
  v8h a1h = *(const v8h*)(hrow + 40 + koff);
  v16h A0 = __builtin_shufflevector(a0l, a0h, 0, 1, 2, 3, 4, 5, 6, 7, 8, 9, 10, 11, 12, 13, 14, 15);
  v16h A1 = __builtin_shufflevector(a1l, a1h, 0, 1, 2, 3, 4, 5, 6, 7, 8, 9, 10, 11, 12, 13, 14, 15);

  v8f acc[12];
#pragma unroll
  for (int ct = 0; ct < 12; ++ct) {
    const _Float16* bp = wlds + ((ct * 2 + 0) * 32 + lane) * 16;
    v16h B0 = *(const v16h*)bp;
    v16h B1 = *(const v16h*)(bp + 32 * 16);
    v8f c = {0.f, 0.f, 0.f, 0.f, 0.f, 0.f, 0.f, 0.f};
    c = __builtin_amdgcn_wmma_f32_16x16x32_f16(false, A0, false, B0, (short)0, c, false, false);
    c = __builtin_amdgcn_wmma_f32_16x16x32_f16(false, A1, false, B1, (short)0, c, false, false);
    acc[ct] = c;
  }

  // --- fused gate epilogue. C/D layout: vgpr r, lane l: M = r + 8*(l/16), N = l%16.
  // column tiles 0-3 = r-gate, 4-7 = z-gate, 8-11 = n-gate (same lane/vgpr slot).
  const int lsel = lane >> 4;
  const int jc = lane & 15;
#pragma unroll
  for (int ct = 0; ct < 4; ++ct) {
    const int j = ct * 16 + jc;
    const float bir = b_ih[j], biz = b_ih[HID + j], bin_ = b_ih[2 * HID + j];
    const float bhr = b_hh[j], bhz = b_hh[HID + j], bhn = b_hh[2 * HID + j];
    float wr[D], wz[D], wn[D];
#pragma unroll
    for (int d = 0; d < D; ++d) {
      wr[d] = w_ih[j * D + d];
      wz[d] = w_ih[(HID + j) * D + d];
      wn[d] = w_ih[(2 * HID + j) * D + d];
    }
#pragma unroll
    for (int r = 0; r < 8; ++r) {
      const int m = m0 + lsel * 8 + r;
      float xv[D];
      if constexpr (ENC) {
        xv[0] = xin[m];
        const int b = m / NN, n = m % NN;
        xv[1] = pm25[(b * THIST + t) * NN + n];
      } else {
#pragma unroll
        for (int d = 0; d < D; ++d) xv[d] = xin[m * D + d];
      }
      float gir = bir, giz = biz, gin = bin_;
#pragma unroll
      for (int d = 0; d < D; ++d) {
        gir += xv[d] * wr[d];
        giz += xv[d] * wz[d];
        gin += xv[d] * wn[d];
      }
      const float ghr = acc[ct][r] + bhr;
      const float ghz = acc[ct + 4][r] + bhz;
      const float ghn = acc[ct + 8][r] + bhn;
      const float rg = sigmoidf_(gir + ghr);
      const float zg = sigmoidf_(giz + ghz);
      const float ng = tanhf(gin + rg * ghn);
      const float hold = h32_in[m * HID + j];
      const float hnew = (1.0f - zg) * ng + zg * hold;
      h32_out[m * HID + j] = hnew;
      h16_out[m * HID + j] = (_Float16)hnew;
      if constexpr (ENC) {
        const int b = m / NN, n = m % NN;
        hist[(((size_t)b * THIST + t) * NN + n) * HID + j] = (_Float16)hnew;
      }
    }
  }
}

// ---------------------------------------------------------------- fc (encoder)
__global__ void fc_kernel(const float* __restrict__ h, const float* __restrict__ fw,
                          const float* __restrict__ fb, float* __restrict__ xn) {
  int m = blockIdx.x * blockDim.x + threadIdx.x;
  if (m >= NODES) return;
  float s = fb[0];
  const float* hp = h + m * HID;
#pragma unroll 8
  for (int j = 0; j < HID; ++j) s += fw[j] * hp[j];
  xn[m] = s;
}

// ---------------------------------------------------------------- attention + out
// One wave per node: lanes 0-23 compute energy[t], wave softmax, then lanes
// parallelize the 64 hidden dims for the weighted sum + output projection.
__global__ __launch_bounds__(128) void attn_kernel(const float* __restrict__ h32,
                                                   const _Float16* __restrict__ hist,
                                                   const float* __restrict__ out_w,
                                                   const float* __restrict__ out_b,
                                                   float* __restrict__ xn,
                                                   float* __restrict__ preds, int tstep) {
  __shared__ float alpha_s[4][THIST];
  const int lane = threadIdx.x & 31;
  const int wv = threadIdx.x >> 5;
  const int m = blockIdx.x * 4 + wv;
  const int b = m / NN, n = m % NN;
  const float* x = h32 + m * HID;
  const _Float16* Hn = hist + (((size_t)b * THIST) * NN + n) * HID;

  // pull this node's history rows toward the WGP (global_prefetch_b8)
  if (lane < THIST) __builtin_prefetch((const void*)(Hn + (size_t)lane * NN * HID), 0, 0);

  float e = -3.0e38f;
  if (lane < THIST) {
    const _Float16* Ht = Hn + (size_t)lane * NN * HID;
    float s = 0.0f;
#pragma unroll
    for (int d = 0; d < HID; ++d) s += (float)Ht[d] * x[d];
    e = s;
  }
  float mx = e;
#pragma unroll
  for (int o = 16; o > 0; o >>= 1) mx = fmaxf(mx, __shfl_xor(mx, o, 32));
  float ex = (lane < THIST) ? __expf(e - mx) : 0.0f;
  float sm = ex;
#pragma unroll
  for (int o = 16; o > 0; o >>= 1) sm += __shfl_xor(sm, o, 32);
  if (lane < THIST) alpha_s[wv][lane] = ex / sm;
  __syncthreads();

  float acc = 0.0f;
#pragma unroll
  for (int p = 0; p < 2; ++p) {
    const int d = lane + p * 32;
    float ad = 0.0f;
    for (int tt = 0; tt < THIST; ++tt)
      ad += alpha_s[wv][tt] * (float)Hn[(size_t)tt * NN * HID + d];
    acc += out_w[d] * ad + out_w[HID + d] * x[d];
  }
#pragma unroll
  for (int o = 16; o > 0; o >>= 1) acc += __shfl_xor(acc, o, 32);
  if (lane == 0) {
    float o = acc + out_b[0];
    xn[m] = o;
    preds[((size_t)b * TFCST + tstep) * NN + n] = o;
  }
}

// ---------------------------------------------------------------- launcher
extern "C" void kernel_launch(void* const* d_in, const int* in_sizes, int n_in,
                              void* d_out, int out_size, void* d_ws, size_t ws_size,
                              hipStream_t stream) {
  (void)in_sizes; (void)n_in; (void)out_size; (void)ws_size;
  const float* pm25   = (const float*)d_in[0];
  const float* feat   = (const float*)d_in[1];
  const int*   ei     = (const int*)d_in[2];
  const float* w_ih_e = (const float*)d_in[3];
  const float* w_hh_e = (const float*)d_in[4];
  const float* b_ih_e = (const float*)d_in[5];
  const float* b_hh_e = (const float*)d_in[6];
  const float* fc_w   = (const float*)d_in[7];
  const float* fc_b   = (const float*)d_in[8];
  const float* cw0    = (const float*)d_in[9];
  const float* cw1    = (const float*)d_in[10];
  const float* cb     = (const float*)d_in[11];
  const float* w_ih_d = (const float*)d_in[12];
  const float* w_hh_d = (const float*)d_in[13];
  const float* b_ih_d = (const float*)d_in[14];
  const float* b_hh_d = (const float*)d_in[15];
  const float* out_w  = (const float*)d_in[16];
  const float* out_b  = (const float*)d_in[17];
  float* out = (float*)d_out;

  uintptr_t base = (uintptr_t)d_ws;
  size_t off = 0;
  auto carve = [&](size_t bytes) -> void* {
    void* p = (void*)(base + off);
    off += (bytes + 255) & ~(size_t)255;
    return p;
  };
  float*     h32[2] = {(float*)carve((size_t)NODES * HID * 4), (float*)carve((size_t)NODES * HID * 4)};
  _Float16*  h16[2] = {(_Float16*)carve((size_t)NODES * HID * 2), (_Float16*)carve((size_t)NODES * HID * 2)};
  _Float16*  hist   = (_Float16*)carve((size_t)NB * THIST * NN * HID * 2);  // 98.3 MB: fits L2 at fp16
  float*     xn     = (float*)carve((size_t)NODES * 4);
  float*     tx1    = (float*)carve((size_t)NODES * 9 * 4);
  float*     xb     = (float*)carve((size_t)NODES * 9 * 4);
  float*     x2     = (float*)carve((size_t)NODES * 10 * 4);
  float*     deg    = (float*)carve((size_t)NODES * 4);
  float*     nrm    = (float*)carve((size_t)NEDGE * 4);
  _Float16*  pkE    = (_Float16*)carve(WPACK_HALFS * 2);
  _Float16*  pkD    = (_Float16*)carve(WPACK_HALFS * 2);

  // init state + graph preprocessing (once per call; deterministic)
  zero_f32_kernel<<<(NODES * HID + 255) / 256, 256, 0, stream>>>(h32[0], NODES * HID);
  zero_f32_kernel<<<(NODES * 32 + 255) / 256, 256, 0, stream>>>((float*)h16[0], NODES * 32);
  zero_f32_kernel<<<(NODES + 255) / 256, 256, 0, stream>>>(xn, NODES);
  zero_f32_kernel<<<(NODES + 255) / 256, 256, 0, stream>>>(deg, NODES);
  pack_whh_kernel<<<WPACK_HALFS / 256, 256, 0, stream>>>(w_hh_e, pkE);
  pack_whh_kernel<<<WPACK_HALFS / 256, 256, 0, stream>>>(w_hh_d, pkD);
  deg_kernel<<<NEDGE / 256, 256, 0, stream>>>(ei, deg);
  dis_kernel<<<(NODES + 255) / 256, 256, 0, stream>>>(deg);
  norm_kernel<<<NEDGE / 256, 256, 0, stream>>>(ei, deg, nrm);

  int cur = 0;
  // ---------------- encoder ----------------
  for (int t = 0; t < THIST; ++t) {
    gru_step_kernel<2, true><<<NODES / 64, 128, 0, stream>>>(
        h16[cur], h32[cur], h32[cur ^ 1], h16[cur ^ 1], xn, pm25, t, pkE,
        w_ih_e, b_ih_e, b_hh_e, hist);
    fc_kernel<<<(NODES + 127) / 128, 128, 0, stream>>>(h32[cur ^ 1], fc_w, fc_b, xn);
    cur ^= 1;
  }
  // ---------------- decoder ----------------
  for (int t = 0; t < TFCST; ++t) {
    zero_f32_kernel<<<(NODES * 9 + 255) / 256, 256, 0, stream>>>(tx1, NODES * 9);
    build_x_kernel<<<(NODES + 255) / 256, 256, 0, stream>>>(xn, feat, t, xb);
    cheb_scatter_kernel<<<NEDGE / 256, 256, 0, stream>>>(ei, nrm, xb, tx1);
    cheb_finish_kernel<<<(NODES + 255) / 256, 256, 0, stream>>>(xb, tx1, cw0, cw1, cb, x2);
    gru_step_kernel<10, false><<<NODES / 64, 128, 0, stream>>>(
        h16[cur], h32[cur], h32[cur ^ 1], h16[cur ^ 1], x2, nullptr, t, pkD,
        w_ih_d, b_ih_d, b_hh_d, nullptr);
    attn_kernel<<<NODES / 4, 128, 0, stream>>>(h32[cur ^ 1], hist, out_w, out_b, xn, out, t);
    cur ^= 1;
  }
}
